// GlobalFeatureExtractor_11639361372345
// MI455X (gfx1250) — matrix operand order repaired and checked by reading, not compiled, not run on hardware
//
#include <hip/hip_runtime.h>
#include <hip/hip_bf16.h>
#include <math.h>

typedef __attribute__((ext_vector_type(16))) _Float16 v16h;
typedef __attribute__((ext_vector_type(8)))  _Float16 v8h;
typedef __attribute__((ext_vector_type(4)))  _Float16 v4h;
typedef __attribute__((ext_vector_type(8)))  float    v8f;
typedef __attribute__((ext_vector_type(4)))  unsigned u32x4;
typedef __attribute__((ext_vector_type(8)))  unsigned u32x8;

#define BN_EPS 1e-5f
#define LN_EPS 1e-5f
#define USE_TDM 1
static const int NIMG = 8;

__device__ __forceinline__ int imin(int a, int b) { return a < b ? a : b; }

// ============================================================================
// WMMA GEMM:  C[m,n] = act( bn( sum_k A[m,k]*B[k,n] ) + bias + resid )
//   A: f32 weights, row-major M x K.
//   BL=0: B is NCHW activation  (addr = (img*K + k)*HW + p,  n = img*HW + p)
//   BL=1: B is row-major        (addr = n*K + k)
//   CL=0: store NCHW            (addr = (img*M + m)*HW + p)
//   CL=1: store row-major       (addr = n*ldc + m)
// Block = 128 threads = 4 waves; block tile 32(M) x 64(N); K stepped by 32.
// BL=1: B tile moved by the Tensor Data Mover (tensor_load_to_lds, f32,
//       32(k) x 64(n) tile, 16B pad per 128B row -> 36-dword k-minor rows).
// BL=0: manual transpose staging (contiguous axis is n), branch-free via
//       coordinate clamping (all K/N quads are all-valid-or-all-invalid).
// ============================================================================
template<int BL, int CL>
__global__ void __launch_bounds__(128)
gemm_wmma(const float* __restrict__ A, const float* __restrict__ B,
          float* __restrict__ C, const float* __restrict__ resid,
          const float* __restrict__ bng, const float* __restrict__ bnb,
          const float* __restrict__ bnm, const float* __restrict__ bnv,
          const float* __restrict__ bias,
          int M, int N, int K, int HW, int ldc, int act)
{
    const int LDA = 40;                      // f16 per A row (80B, 16B-aligned)
    const int LDBF = 36;                     // f32 per TDM B row (144B)
    __shared__ __align__(16) _Float16 As[32 * LDA];
    __shared__ __align__(16) unsigned char BsRaw[64 * LDBF * 4 + 64];
    _Float16* Bs16 = (_Float16*)BsRaw;       // BL==0 view: [64][40] f16
    float*    Bs32 = (float*)BsRaw;          // BL==1 view: [64][36] f32

    const int tid   = threadIdx.x;
    const int lane  = tid & 31;
    const int wv    = tid >> 5;        // wave 0..3 -> N sub-tile
    const int hl    = (lane >> 4) & 1; // half-wave
    const int ml    = lane & 15;
    const int mBase = blockIdx.y * 32;
    const int nBase = blockIdx.x * 64;

    v8f acc0 = {0.f, 0.f, 0.f, 0.f, 0.f, 0.f, 0.f, 0.f};
    v8f acc1 = {0.f, 0.f, 0.f, 0.f, 0.f, 0.f, 0.f, 0.f};

    const int KT = (K + 31) / 32;
    for (int kt = 0; kt < KT; ++kt) {
        const int k0 = kt * 32;

        // ---- stage A tile (32 x 32): clamped vec4 loads, packed v4h stores --
        #pragma unroll
        for (int it = 0; it < 2; ++it) {
            int t  = tid + it * 128;
            int r  = t >> 3;
            int kq = (t & 7) * 4;
            int gm = mBase + r, gk = k0 + kq;
            bool ok = (gm < M) && (gk < K);
            int cm = imin(gm, M - 1);
            int ck = imin(gk, K - 4);
            float4 f = *(const float4*)(A + (size_t)cm * K + ck);
            v4h h;
            h[0] = ok ? (_Float16)f.x : (_Float16)0.f;
            h[1] = ok ? (_Float16)f.y : (_Float16)0.f;
            h[2] = ok ? (_Float16)f.z : (_Float16)0.f;
            h[3] = ok ? (_Float16)f.w : (_Float16)0.f;
            *(v4h*)&As[r * LDA + kq] = h;
        }

        if (BL == 1) {
#if USE_TDM
            // ---- Tensor Data Mover: one DMA per block loads the B tile ----
            if (wv == 0) {
                unsigned long long ga =
                    (unsigned long long)(const void*)(B + ((size_t)nBase * K + k0));
                unsigned lds = (unsigned)(size_t)(void*)Bs32;
                unsigned td0 = (unsigned)(K - k0);   // remaining k (OOB -> 0)
                unsigned td1 = (unsigned)(N - nBase);// remaining n (OOB -> 0)
                u32x4 g0;
                g0[0] = 1u;                              // count=1, user mode
                g0[1] = lds;                             // lds_addr
                g0[2] = (unsigned)ga;                    // global_addr[31:0]
                g0[3] = (unsigned)((ga >> 32) & 0x1FFFFFFu) | (2u << 30); // type=2
                u32x8 g1;
                // data_size=4B(code2), pad_enable, pad_interval=32dw(code4),
                // pad_amount=4dw(code3); workgroup_mask=0
                g1[0] = (2u << 16) | (1u << 20) | (4u << 22) | (3u << 25);
                g1[1] = (td0 & 0xFFFFu) << 16;           // tensor_dim0[15:0]
                g1[2] = (td0 >> 16) | ((td1 & 0xFFFFu) << 16);
                g1[3] = (td1 >> 16) | (32u << 16);       // tile_dim0 = 32 (k)
                g1[4] = 64u;                             // tile_dim1 = 64 (n)
                g1[5] = (unsigned)K;                     // dim0 stride low
                g1[6] = 0u;                              // stride hi / dim1 stride lo
                g1[7] = 0u;
                asm volatile("tensor_load_to_lds %0, %1"
                             :: "s"(g0), "s"(g1) : "memory");
                __builtin_amdgcn_s_wait_tensorcnt(0);
            }
#else
            #pragma unroll
            for (int it = 0; it < 4; ++it) {
                int t  = tid + it * 128;
                int n  = t >> 3;
                int kq = (t & 7) * 4;
                int gn = nBase + n, gk = k0 + kq;
                bool ok = (gn < N) && (gk < K);
                int cn = imin(gn, N - 1);
                int ck = imin(gk, K - 4);
                float4 f = *(const float4*)(B + (size_t)cn * K + ck);
                float4 z = {0.f, 0.f, 0.f, 0.f};
                if (!ok) f = z;
                Bs32[n * LDBF + kq + 0] = f.x;
                Bs32[n * LDBF + kq + 1] = f.y;
                Bs32[n * LDBF + kq + 2] = f.z;
                Bs32[n * LDBF + kq + 3] = f.w;
            }
#endif
        } else {
            // NCHW source: vec4 along n (contiguous pixels), transpose to
            // K-minor f16 tile; clamped unconditional loads.
            #pragma unroll
            for (int it = 0; it < 4; ++it) {
                int t  = tid + it * 128;
                int k  = t >> 4;
                int nq = (t & 15) * 4;
                int gk = k0 + k, gn = nBase + nq;
                bool ok = (gk < K) && (gn < N);
                int ck = imin(gk, K - 1);
                int cn = imin(gn, N - 4);
                int img = cn / HW, p = cn - img * HW;
                float4 f = *(const float4*)(B + ((size_t)img * K + ck) * HW + p);
                Bs16[(nq + 0) * LDA + k] = ok ? (_Float16)f.x : (_Float16)0.f;
                Bs16[(nq + 1) * LDA + k] = ok ? (_Float16)f.y : (_Float16)0.f;
                Bs16[(nq + 2) * LDA + k] = ok ? (_Float16)f.z : (_Float16)0.f;
                Bs16[(nq + 3) * LDA + k] = ok ? (_Float16)f.w : (_Float16)0.f;
            }
        }
        if (kt + 1 < KT)   // hint next A tile (global_prefetch_b8)
            __builtin_prefetch(&A[(size_t)mBase * K + (k0 + 32)], 0, 1);
        __syncthreads();

        // ---- fragments: contiguous b128 LDS loads in contiguous k order (16*hl+s) -
        // A lane: k run  [16*hl..+15] of row m.
        // B lane: k run  [16*hl..+15] of row n.
        const _Float16* a0p = &As[ml * LDA];
        const _Float16* a1p = &As[(16 + ml) * LDA];
        v8h a0lo = *(const v8h*)(a0p + 16 * hl);
        v8h a0hi = *(const v8h*)(a0p + 16 * hl + 8);
        v8h a1lo = *(const v8h*)(a1p + 16 * hl);
        v8h a1hi = *(const v8h*)(a1p + 16 * hl + 8);
        v16h af0, af1, bfr;
        if (BL == 1) {
            const float* bp = Bs32 + (wv * 16 + ml) * LDBF + 16 * hl;
            float4 b0 = *(const float4*)(bp + 0);
            float4 b1 = *(const float4*)(bp + 4);
            float4 b2 = *(const float4*)(bp + 8);
            float4 b3 = *(const float4*)(bp + 12);
            bfr[0]  = (_Float16)b0.x; bfr[1]  = (_Float16)b0.y;
            bfr[2]  = (_Float16)b0.z; bfr[3]  = (_Float16)b0.w;
            bfr[4]  = (_Float16)b1.x; bfr[5]  = (_Float16)b1.y;
            bfr[6]  = (_Float16)b1.z; bfr[7]  = (_Float16)b1.w;
            bfr[8]  = (_Float16)b2.x; bfr[9]  = (_Float16)b2.y;
            bfr[10] = (_Float16)b2.z; bfr[11] = (_Float16)b2.w;
            bfr[12] = (_Float16)b3.x; bfr[13] = (_Float16)b3.y;
            bfr[14] = (_Float16)b3.z; bfr[15] = (_Float16)b3.w;
        } else {
            const _Float16* bp = &Bs16[(wv * 16 + ml) * LDA];
            v8h blo = *(const v8h*)(bp + 16 * hl);
            v8h bhi = *(const v8h*)(bp + 16 * hl + 8);
            #pragma unroll
            for (int i = 0; i < 8; ++i) { bfr[i] = blo[i]; bfr[8 + i] = bhi[i]; }
        }
        #pragma unroll
        for (int i = 0; i < 8; ++i) {
            af0[i] = a0lo[i]; af0[8 + i] = a0hi[i];
            af1[i] = a1lo[i]; af1[8 + i] = a1hi[i];
        }
        acc0 = __builtin_amdgcn_wmma_f32_16x16x32_f16(
                   false, af0, false, bfr, (short)0, acc0, false, false);
        acc1 = __builtin_amdgcn_wmma_f32_16x16x32_f16(
                   false, af1, false, bfr, (short)0, acc1, false, false);
        __syncthreads();
    }

    // ---- fused epilogue: BN, bias, residual, ReLU, layout-aware store ----
    const int gn = nBase + wv * 16 + ml;
    int img = 0, p = 0;
    if (CL == 0) { img = gn / HW; p = gn - img * HW; }
    #pragma unroll
    for (int t = 0; t < 2; ++t) {
        const v8f& acc = t ? acc1 : acc0;
        #pragma unroll
        for (int r = 0; r < 8; ++r) {
            int gm = mBase + 16 * t + r + 8 * hl;
            if (gm < M && gn < N) {
                float val = acc[r];
                if (bng) {
                    float inv = rsqrtf(bnv[gm] + BN_EPS);
                    float sc  = bng[gm] * inv;
                    val = val * sc + (bnb[gm] - bnm[gm] * sc);
                }
                if (bias) val += bias[gm];
                size_t oidx;
                if (CL == 0) oidx = ((size_t)img * M + gm) * HW + p;
                else         oidx = (size_t)gn * ldc + gm;
                if (resid) val += resid[oidx];
                if (act == 1) val = fmaxf(val, 0.f);
                C[oidx] = val;
            }
        }
    }
}

// ============================================================================
// Depthwise 3x3 conv (pad 1, stride 1/2) + fused BN + ReLU, NCHW.
// ============================================================================
__global__ void __launch_bounds__(256)
dwconv3x3_bn_relu(const float* __restrict__ x, const float* __restrict__ w,
                  const float* __restrict__ bng, const float* __restrict__ bnb,
                  const float* __restrict__ bnm, const float* __restrict__ bnv,
                  float* __restrict__ y,
                  int C, int H, int W, int Ho, int Wo, int stride, int NC)
{
    size_t idx = (size_t)blockIdx.x * blockDim.x + threadIdx.x;
    size_t total = (size_t)NC * Ho * Wo;
    if (idx >= total) return;
    int wo = idx % Wo;
    size_t t = idx / Wo;
    int ho = t % Ho;
    int nc = t / Ho;
    int c  = nc % C;
    const float* xp = x + (size_t)nc * H * W;
    const float* wp = w + c * 9;
    float s = 0.f;
    #pragma unroll
    for (int kh = 0; kh < 3; ++kh) {
        int hi = ho * stride - 1 + kh;
        if (hi < 0 || hi >= H) continue;
        #pragma unroll
        for (int kw = 0; kw < 3; ++kw) {
            int wi = wo * stride - 1 + kw;
            if (wi < 0 || wi >= W) continue;
            s += wp[kh * 3 + kw] * xp[(size_t)hi * W + wi];
        }
    }
    float inv = rsqrtf(bnv[c] + BN_EPS);
    float sc  = bng[c] * inv;
    s = s * sc + (bnb[c] - bnm[c] * sc);
    y[idx] = fmaxf(s, 0.f);
}

// ============================================================================
// PVM pre: NCHW -> (B, L, C) with layernorm over C.
// ============================================================================
__global__ void __launch_bounds__(256)
ln_nchw_to_blc(const float* __restrict__ x, const float* __restrict__ g,
               const float* __restrict__ b, float* __restrict__ xn,
               int Bn, int C, int L)
{
    size_t idx = (size_t)blockIdx.x * blockDim.x + threadIdx.x;
    if (idx >= (size_t)Bn * L) return;
    int bimg = idx / L;
    int l    = idx % L;
    const float* xp = x + (size_t)bimg * C * L + l;   // stride L per channel
    float mu = 0.f;
    for (int c = 0; c < C; ++c) mu += xp[(size_t)c * L];
    mu /= C;
    float var = 0.f;
    for (int c = 0; c < C; ++c) { float d = xp[(size_t)c * L] - mu; var += d * d; }
    var /= C;
    float rs = rsqrtf(var + LN_EPS);
    float* op = xn + idx * (size_t)C;
    for (int c = 0; c < C; ++c)
        op[c] = (xp[(size_t)c * L] - mu) * rs * g[c] + b[c];
}

// (B, L, C) -> chunked (4B, L, dm) with b4 = g*B + b
__global__ void __launch_bounds__(256)
chunk_copy(const float* __restrict__ xn, float* __restrict__ xch,
           int Bn, int L, int C, int dm)
{
    size_t idx = (size_t)blockIdx.x * blockDim.x + threadIdx.x;
    size_t total = (size_t)4 * Bn * L * dm;
    if (idx >= total) return;
    int j = idx % dm;
    size_t t = idx / dm;
    int l  = t % L;
    int b4 = t / L;
    int g    = b4 / Bn;
    int bimg = b4 - g * Bn;
    xch[idx] = xn[((size_t)bimg * L + l) * C + g * dm + j];
}

// Causal depthwise conv1d (k=4, left pad 3) over xz[..., :di] + bias + SiLU.
__global__ void __launch_bounds__(256)
conv1d_silu(const float* __restrict__ xz, const float* __restrict__ cw,
            const float* __restrict__ cb, float* __restrict__ xc,
            int B4, int L, int di)
{
    size_t idx = (size_t)blockIdx.x * blockDim.x + threadIdx.x;
    size_t total = (size_t)B4 * L * di;
    if (idx >= total) return;
    int d = idx % di;
    size_t t = idx / di;
    int l  = t % L;
    int b4 = t / L;
    const float* xp = xz + (size_t)b4 * L * 2 * di + d;   // + l*2*di per row
    float s = cb[d];
    #pragma unroll
    for (int k = 0; k < 4; ++k) {
        int li = l - 3 + k;
        if (li >= 0) s += cw[d * 4 + k] * xp[(size_t)li * 2 * di];
    }
    xc[idx] = s / (1.f + expf(-s));   // silu
}

// dt = softplus(dbl[:, :dtr] @ dt_w^T + dt_b)
__global__ void __launch_bounds__(256)
dt_softplus(const float* __restrict__ dbl, const float* __restrict__ dtw,
            const float* __restrict__ dtb, float* __restrict__ dt,
            int ntot, int di, int dtr, int ldd)
{
    size_t idx = (size_t)blockIdx.x * blockDim.x + threadIdx.x;
    if (idx >= (size_t)ntot * di) return;
    int d = idx % di;
    size_t n = idx / di;
    float s = dtb[d];
    for (int r = 0; r < dtr; ++r) s += dbl[n * ldd + r] * dtw[d * dtr + r];
    dt[idx] = (s > 20.f) ? s : log1pf(expf(s));
}

// Selective scan: one thread per (sequence b4, channel d); 16-state recurrence.
__global__ void __launch_bounds__(256)
ssm_scan(const float* __restrict__ dt, const float* __restrict__ dbl,
         const float* __restrict__ xc, const float* __restrict__ A_log,
         float* __restrict__ ys, int B4, int L, int di, int dtr, int ldd)
{
    int t = blockIdx.x * blockDim.x + threadIdx.x;
    if (t >= B4 * di) return;
    int b4 = t / di;
    int d  = t - b4 * di;
    float Ad[16];
    #pragma unroll
    for (int s = 0; s < 16; ++s) Ad[s] = -expf(A_log[d * 16 + s]);
    float h[16];
    #pragma unroll
    for (int s = 0; s < 16; ++s) h[s] = 0.f;
    for (int l = 0; l < L; ++l) {
        size_t base = (size_t)b4 * L + l;
        float dtv = dt[base * di + d];
        float xv  = xc[base * di + d];
        const float* Bv = dbl + base * ldd + dtr;
        const float* Cv = Bv + 16;
        float y = 0.f;
        #pragma unroll
        for (int s = 0; s < 16; ++s) {
            h[s] = h[s] * expf(dtv * Ad[s]) + (dtv * xv) * Bv[s];
            y += h[s] * Cv[s];
        }
        ys[base * di + d] = y;
    }
}

// y2 = (ys + xc*D) * silu(z),  z = xz[..., di:]
__global__ void __launch_bounds__(256)
gate_mul(const float* __restrict__ ys, const float* __restrict__ xc,
         const float* __restrict__ xz, const float* __restrict__ Dp,
         float* __restrict__ y2, size_t total, int di)
{
    size_t idx = (size_t)blockIdx.x * blockDim.x + threadIdx.x;
    if (idx >= total) return;
    int d = idx % di;
    size_t n = idx / di;
    float z = xz[n * 2 * di + di + d];
    y2[idx] = (ys[idx] + xc[idx] * Dp[d]) * (z / (1.f + expf(-z)));
}

// PVM mid: reassemble chunks, add skip*xn, layernorm over C -> tn (B,L,C)
__global__ void __launch_bounds__(256)
pvm_mid_ln(const float* __restrict__ ym, const float* __restrict__ xn,
           const float* __restrict__ skip, const float* __restrict__ g,
           const float* __restrict__ b, float* __restrict__ tn,
           int Bn, int L, int C, int dm)
{
    size_t idx = (size_t)blockIdx.x * blockDim.x + threadIdx.x;
    if (idx >= (size_t)Bn * L) return;
    int bimg = idx / L;
    int l    = idx % L;
    float sk = skip[0];
    float mu = 0.f, var = 0.f;
    for (int c = 0; c < C; ++c) {
        int gch = c / dm, j = c - gch * dm;
        float v = ym[(((size_t)(gch * Bn + bimg)) * L + l) * dm + j]
                + sk * xn[idx * (size_t)C + c];
        mu += v;
    }
    mu /= C;
    for (int c = 0; c < C; ++c) {
        int gch = c / dm, j = c - gch * dm;
        float v = ym[(((size_t)(gch * Bn + bimg)) * L + l) * dm + j]
                + sk * xn[idx * (size_t)C + c];
        float d = v - mu; var += d * d;
    }
    var /= C;
    float rs = rsqrtf(var + LN_EPS);
    for (int c = 0; c < C; ++c) {
        int gch = c / dm, j = c - gch * dm;
        float v = ym[(((size_t)(gch * Bn + bimg)) * L + l) * dm + j]
                + sk * xn[idx * (size_t)C + c];
        tn[idx * (size_t)C + c] = (v - mu) * rs * g[c] + b[c];
    }
}

// ============================================================================
// SE: global average pool, tiny MLP + sigmoid, channel scale.
// ============================================================================
__global__ void __launch_bounds__(256)
gap_mean(const float* __restrict__ x, float* __restrict__ pooled, int HW)
{
    __shared__ float red[256];
    int nc = blockIdx.x;
    const float* xp = x + (size_t)nc * HW;
    float s = 0.f;
    for (int p = threadIdx.x; p < HW; p += 256) s += xp[p];
    red[threadIdx.x] = s;
    __syncthreads();
    for (int off = 128; off > 0; off >>= 1) {
        if (threadIdx.x < off) red[threadIdx.x] += red[threadIdx.x + off];
        __syncthreads();
    }
    if (threadIdx.x == 0) pooled[nc] = red[0] / (float)HW;
}

__global__ void __launch_bounds__(128)
se_mlp(const float* __restrict__ pooled, const float* __restrict__ w1,
       const float* __restrict__ w2, float* __restrict__ gate, int C)
{
    __shared__ float hid[8];
    int n = blockIdx.x;
    int hn = C / 16;
    if ((int)threadIdx.x < hn) {
        float s = 0.f;
        for (int c = 0; c < C; ++c) s += pooled[n * C + c] * w1[threadIdx.x * C + c];
        hid[threadIdx.x] = fmaxf(s, 0.f);
    }
    __syncthreads();
    if ((int)threadIdx.x < C) {
        float s = 0.f;
        for (int r = 0; r < hn; ++r) s += hid[r] * w2[threadIdx.x * hn + r];
        gate[n * C + threadIdx.x] = 1.f / (1.f + expf(-s));
    }
}

__global__ void __launch_bounds__(256)
se_scale(const float* __restrict__ src, const float* __restrict__ gate,
         float* __restrict__ dst, size_t total, int HW)
{
    size_t idx = (size_t)blockIdx.x * blockDim.x + threadIdx.x;
    if (idx >= total) return;
    dst[idx] = src[idx] * gate[idx / HW];
}

// ============================================================================
// Host side: parameter parsing, workspace carving, pipeline.
// ============================================================================
struct BNp  { const float *g, *b, *m, *v; };
struct LBp  { const float* w1; BNp bn1; const float* wd; BNp bn2; const float* w2; BNp bn3; };
struct Mbp  { const float *in_w, *conv_w, *conv_b, *xproj_w, *dt_w, *dt_b, *A_log, *D, *out_w; };
struct PVMp { const float *ln_g, *ln_b; Mbp m; const float *proj_w, *proj_b, *skip; };
struct SEp  { const float *w1, *w2; };
struct Stagep { LBp lb0, lb1, lb3; PVMp pv2, pv4; };

struct Cur { void* const* in; int i;
             const float* n() { return (const float*)in[i++]; } };
static BNp  rdBN(Cur& c)  { BNp r; r.g=c.n(); r.b=c.n(); r.m=c.n(); r.v=c.n(); return r; }
static LBp  rdLB(Cur& c)  { LBp r; r.w1=c.n(); r.bn1=rdBN(c); r.wd=c.n(); r.bn2=rdBN(c);
                            r.w2=c.n(); r.bn3=rdBN(c); return r; }
static Mbp  rdMb(Cur& c)  { Mbp r; r.in_w=c.n(); r.conv_w=c.n(); r.conv_b=c.n();
                            r.xproj_w=c.n(); r.dt_w=c.n(); r.dt_b=c.n();
                            r.A_log=c.n(); r.D=c.n(); r.out_w=c.n(); return r; }
static PVMp rdPVM(Cur& c) { PVMp r; r.ln_g=c.n(); r.ln_b=c.n(); r.m=rdMb(c);
                            r.proj_w=c.n(); r.proj_b=c.n(); r.skip=c.n(); return r; }
static Stagep rdStage(Cur& c) { Stagep r; r.lb0=rdLB(c); r.lb1=rdLB(c);
                                r.pv2=rdPVM(c); r.lb3=rdLB(c); r.pv4=rdPVM(c); return r; }

struct WSAlloc { char* p; size_t off;
    float* a(size_t elems) {
        float* r = (float*)(p + off);
        off += ((elems * sizeof(float)) + 255) & ~(size_t)255;
        return r;
    } };

struct Bufs { float *bufP, *bufQ, *e1, *e2;
              float *xn, *xch, *xz, *xc, *dbl, *dt, *ys, *y2, *ym, *tn;
              float *pooled, *gate; };

template<int BL, int CL>
static void launch_gemm(const float* A, const float* B, float* C, const float* resid,
                        const BNp* bn, const float* bias,
                        int M, int N, int K, int HW, int ldc, int act, hipStream_t st)
{
    dim3 grid((N + 63) / 64, (M + 31) / 32);
    const float *g = nullptr, *b = nullptr, *m = nullptr, *v = nullptr;
    if (bn) { g = bn->g; b = bn->b; m = bn->m; v = bn->v; }
    gemm_wmma<BL, CL><<<grid, 128, 0, st>>>(A, B, C, resid, g, b, m, v, bias,
                                            M, N, K, HW, ldc, act);
}

static void run_lb(const LBp& p, const float* xin, float* out,
                   int Cin, int Cout, int H, int W, int stride, bool shortcut,
                   Bufs& bf, hipStream_t st)
{
    const int ce = Cin * 6;
    const int Ho = (stride == 2) ? H / 2 : H;
    const int Wo = (stride == 2) ? W / 2 : W;
    // expand 1x1 + BN + ReLU (WMMA)
    launch_gemm<0, 0>(p.w1, xin, bf.e1, nullptr, &p.bn1, nullptr,
                      ce, NIMG * H * W, Cin, H * W, 0, 1, st);
    // depthwise 3x3 + BN + ReLU
    size_t tdw = (size_t)NIMG * ce * Ho * Wo;
    dwconv3x3_bn_relu<<<(tdw + 255) / 256, 256, 0, st>>>(
        bf.e1, p.wd, p.bn2.g, p.bn2.b, p.bn2.m, p.bn2.v, bf.e2,
        ce, H, W, Ho, Wo, stride, NIMG * ce);
    // project 1x1 + BN (+ residual) (WMMA)
    launch_gemm<0, 0>(p.w2, bf.e2, out, shortcut ? xin : nullptr, &p.bn3, nullptr,
                      Cout, NIMG * Ho * Wo, ce, Ho * Wo, 0, 0, st);
}

static void run_pvm(const PVMp& p, const float* xin, float* out,
                    int C, int H, int W, Bufs& bf, hipStream_t st)
{
    const int L = H * W, Bn = NIMG, dm = C / 4, di = 2 * dm;
    const int dtr = (dm + 15) / 16, B4 = 4 * Bn, ldd = 48;
    const size_t totBL = (size_t)Bn * L;
    ln_nchw_to_blc<<<(totBL + 255) / 256, 256, 0, st>>>(xin, p.ln_g, p.ln_b,
                                                        bf.xn, Bn, C, L);
    size_t tch = (size_t)B4 * L * dm;
    chunk_copy<<<(tch + 255) / 256, 256, 0, st>>>(bf.xn, bf.xch, Bn, L, C, dm);
    // in_proj (WMMA, TDM-staged B): xz (B4,L,2di)
    launch_gemm<1, 1>(p.m.in_w, bf.xch, bf.xz, nullptr, nullptr, nullptr,
                      2 * di, B4 * L, dm, 0, 2 * di, 0, st);
    size_t tcd = (size_t)B4 * L * di;
    conv1d_silu<<<(tcd + 255) / 256, 256, 0, st>>>(bf.xz, p.m.conv_w, p.m.conv_b,
                                                   bf.xc, B4, L, di);
    // x_proj (WMMA): dbl (B4,L,dtr+32) with ldc=48
    launch_gemm<1, 1>(p.m.xproj_w, bf.xc, bf.dbl, nullptr, nullptr, nullptr,
                      dtr + 2 * 16, B4 * L, di, 0, ldd, 0, st);
    dt_softplus<<<(tcd + 255) / 256, 256, 0, st>>>(bf.dbl, p.m.dt_w, p.m.dt_b,
                                                   bf.dt, B4 * L, di, dtr, ldd);
    ssm_scan<<<(B4 * di + 255) / 256, 256, 0, st>>>(bf.dt, bf.dbl, bf.xc,
                                                    p.m.A_log, bf.ys,
                                                    B4, L, di, dtr, ldd);
    gate_mul<<<(tcd + 255) / 256, 256, 0, st>>>(bf.ys, bf.xc, bf.xz, p.m.D,
                                                bf.y2, tcd, di);
    // out_proj (WMMA): ym (B4,L,dm)
    launch_gemm<1, 1>(p.m.out_w, bf.y2, bf.ym, nullptr, nullptr, nullptr,
                      dm, B4 * L, di, 0, dm, 0, st);
    pvm_mid_ln<<<(totBL + 255) / 256, 256, 0, st>>>(bf.ym, bf.xn, p.skip,
                                                    p.ln_g, p.ln_b, bf.tn,
                                                    Bn, L, C, dm);
    // projection C x C (WMMA) with bias, stored straight back to NCHW
    launch_gemm<1, 0>(p.proj_w, bf.tn, out, nullptr, nullptr, p.proj_b,
                      C, Bn * L, C, L, 0, 0, st);
}

static void run_se(const SEp& p, const float* src, float* dst,
                   int C, int HW, Bufs& bf, hipStream_t st)
{
    gap_mean<<<NIMG * C, 256, 0, st>>>(src, bf.pooled, HW);
    se_mlp<<<NIMG, 128, 0, st>>>(bf.pooled, p.w1, p.w2, bf.gate, C);
    size_t tot = (size_t)NIMG * C * HW;
    se_scale<<<(tot + 255) / 256, 256, 0, st>>>(src, bf.gate, dst, tot, HW);
}

extern "C" void kernel_launch(void* const* d_in, const int* in_sizes, int n_in,
                              void* d_out, int out_size, void* d_ws, size_t ws_size,
                              hipStream_t stream)
{
    (void)in_sizes; (void)n_in; (void)out_size; (void)ws_size;
    const float* x = (const float*)d_in[0];
    Cur cur{d_in, 1};
    Stagep s1 = rdStage(cur), s2 = rdStage(cur), s3 = rdStage(cur);
    SEp se0{cur.n(), cur.n()}, se1{cur.n(), cur.n()}, se2{cur.n(), cur.n()};

    WSAlloc ws{(char*)d_ws, 0};
    Bufs bf;
    bf.bufP = ws.a((size_t)8 * 64 * 64 * 64);      // 2M elems (max NCHW intermediate)
    bf.bufQ = ws.a((size_t)8 * 64 * 64 * 64);
    bf.e1   = ws.a((size_t)8 * 384 * 128 * 128);   // 50.3M (stage1 expand @128^2)
    bf.e2   = ws.a((size_t)8 * 384 * 64 * 64);     // 12.6M
    bf.xn   = ws.a((size_t)8 * 4096 * 64);
    bf.xch  = ws.a((size_t)32 * 4096 * 16);
    bf.xz   = ws.a((size_t)32 * 4096 * 64);
    bf.xc   = ws.a((size_t)32 * 4096 * 32);
    bf.dbl  = ws.a((size_t)32 * 4096 * 48);
    bf.dt   = ws.a((size_t)32 * 4096 * 32);
    bf.ys   = ws.a((size_t)32 * 4096 * 32);
    bf.y2   = ws.a((size_t)32 * 4096 * 32);
    bf.ym   = ws.a((size_t)32 * 4096 * 16);
    bf.tn   = ws.a((size_t)8 * 4096 * 64);
    bf.pooled = ws.a(8 * 128);
    bf.gate   = ws.a(8 * 128);

    // ----- stage 1: 64 -> 64, 128^2 -> 64^2 -----
    run_lb(s1.lb0, x,       bf.bufP, 64, 64, 128, 128, 2, false, bf, stream);
    run_lb(s1.lb1, bf.bufP, bf.bufQ, 64, 64,  64,  64, 1, true,  bf, stream);
    run_pvm(s1.pv2, bf.bufQ, bf.bufP, 64, 64, 64, bf, stream);
    run_lb(s1.lb3, bf.bufP, bf.bufQ, 64, 64,  64,  64, 1, true,  bf, stream);
    run_pvm(s1.pv4, bf.bufQ, bf.bufP, 64, 64, 64, bf, stream);
    run_se(se0, bf.bufP, bf.bufP, 64, 64 * 64, bf, stream);

    // ----- stage 2: 64 -> 96, 64^2 -> 32^2 -----
    run_lb(s2.lb0, bf.bufP, bf.bufQ, 64, 96, 64, 64, 2, false, bf, stream);
    run_lb(s2.lb1, bf.bufQ, bf.bufP, 96, 96, 32, 32, 1, true,  bf, stream);
    run_pvm(s2.pv2, bf.bufP, bf.bufQ, 96, 32, 32, bf, stream);
    run_lb(s2.lb3, bf.bufQ, bf.bufP, 96, 96, 32, 32, 1, true,  bf, stream);
    run_pvm(s2.pv4, bf.bufP, bf.bufQ, 96, 32, 32, bf, stream);
    run_se(se1, bf.bufQ, bf.bufQ, 96, 32 * 32, bf, stream);

    // ----- stage 3: 96 -> 128, 32^2 -----
    run_lb(s3.lb0, bf.bufQ, bf.bufP, 96, 128, 32, 32, 1, false, bf, stream);
    run_lb(s3.lb1, bf.bufP, bf.bufQ, 128, 128, 32, 32, 1, true, bf, stream);
    run_pvm(s3.pv2, bf.bufQ, bf.bufP, 128, 32, 32, bf, stream);
    run_lb(s3.lb3, bf.bufP, bf.bufQ, 128, 128, 32, 32, 1, true, bf, stream);
    run_pvm(s3.pv4, bf.bufQ, bf.bufP, 128, 32, 32, bf, stream);
    run_se(se2, bf.bufP, (float*)d_out, 128, 32 * 32, bf, stream);
}
